// CompressionAugmentedTrainer_86199993631149
// MI455X (gfx1250) — compile-verified, matrix operand order checked by reading, not run on hardware
//
#include <hip/hip_runtime.h>
#include <hip/hip_bf16.h>
#include <math.h>

// ---------------------------------------------------------------------------
// CompressionAugmentedTrainer for MI455X (gfx1250, wave32, WMMA + TDM)
//
//  1) build_views : per (b,c) row, 1 fwd FFT + 3 masked inv FFTs in LDS,
//     writes 5 views as bf16 A-matrix [320, 262144] (spectral ops are linear
//     => pure bin masks on one spectrum).
//  2) gemm_views  : A[320,262144](bf16) x W[262144,512](fp32->bf16) with
//     v_wmma_f32_16x16x32_bf16. A tile staged to LDS by the Tensor Data
//     Mover (tensor_load_to_lds + s_wait_tensorcnt, D# pad feature produces
//     the padded 40-short LDS row stride). 10 waves/block, each wave owns a
//     fixed 2x4 grid of 16x16 tiles (compile-time indices). Fragments load
//     as two ds_load_b128 each (W staged transposed). Grid = 8 N-tiles x 64
//     K-chunks; W read once from HBM. Deterministic split-K partials.
//  3) reduce_partials (+bias) -> allf[320,512]
//  4) consistency / row-norm / sim+logsumexp / finalize -> scalar loss.
// ---------------------------------------------------------------------------

typedef __attribute__((ext_vector_type(16))) __bf16         v16bf;
typedef __attribute__((ext_vector_type(8)))  unsigned int   v8ui;
typedef __attribute__((ext_vector_type(8)))  float          v8f;
typedef __attribute__((ext_vector_type(4)))  unsigned int   u32x4;
typedef __attribute__((ext_vector_type(4)))  int            i32x4;
typedef __attribute__((ext_vector_type(8)))  int            i32x8;

#define B_SZ   64
#define C_SZ   64
#define T_SZ   4096
#define FEAT   512
#define K_TOT  (C_SZ * T_SZ)          // 262144
#define M_TOT  (5 * B_SZ)             // 320
#define N_TILE 64                     // per-workgroup N
#define K_CHUNK 4096                  // per-workgroup K
#define K_STEP 32                     // per-iteration K (one bf16 WMMA)
#define N_BLKS (FEAT / N_TILE)        // 8
#define K_CHUNKS (K_TOT / K_CHUNK)    // 64
#define LA_STRIDE 40                  // padded LDS row stride (shorts) = 16+4 dwords
#define LB_STRIDE 40                  // padded LDS col stride (shorts)

__device__ __forceinline__ unsigned short f2bf(float f) {
  unsigned int u = __builtin_bit_cast(unsigned int, f);
  u += 0x7FFFu + ((u >> 16) & 1u);    // round-to-nearest-even
  return (unsigned short)(u >> 16);
}

// Load one 16x32 bf16 fragment (A-layout) as two b128 LDS reads:
// lane half h reads k in [8h,8h+8) then [16+8h, 16+8h+8).
__device__ __forceinline__ v16bf load_frag(const unsigned short* base, int h) {
  const uint4 lo = *(const uint4*)(base + 8 * h);
  const uint4 hi = *(const uint4*)(base + 16 + 8 * h);
  v8ui r;
  r[0] = lo.x; r[1] = lo.y; r[2] = lo.z; r[3] = lo.w;
  r[4] = hi.x; r[5] = hi.y; r[6] = hi.z; r[7] = hi.w;
  return __builtin_bit_cast(v16bf, r);
}

// ---------------------------------------------------------------------------
// 4096-point in-place complex FFT in LDS, 512 threads, DIT radix-2.
// ---------------------------------------------------------------------------
__device__ void fft4096(float* re, float* im) {
  const int t = threadIdx.x;
  __syncthreads();
  for (int i = t; i < 4096; i += 512) {   // bit reversal (12 bits)
    int j = (int)(__brev((unsigned)i) >> 20);
    if (i < j) {
      float a = re[i], b = re[j]; re[i] = b; re[j] = a;
      a = im[i]; b = im[j]; im[i] = b; im[j] = a;
    }
  }
  __syncthreads();
  for (int sh = 0; sh < 12; ++sh) {
    const int half = 1 << sh;
    const float ang = -6.283185307179586f / (float)(half << 1);
    for (int k = t; k < 2048; k += 512) {
      const int pos = k & (half - 1);
      const int i0  = ((k >> sh) << (sh + 1)) + pos;
      const int i1  = i0 + half;
      const float ph = ang * (float)pos;
      const float c = __cosf(ph), s = __sinf(ph);
      const float vr = re[i1], vi = im[i1];
      const float tr = vr * c - vi * s;
      const float ti = vr * s + vi * c;
      const float ur = re[i0], ui = im[i0];
      re[i0] = ur + tr; im[i0] = ui + ti;
      re[i1] = ur - tr; im[i1] = ui - ti;
    }
    __syncthreads();
  }
}

__device__ __forceinline__ float blk_reduce_512(float v, float* rbuf) {
  const int t = threadIdx.x;
  rbuf[t] = v; __syncthreads();
  #pragma unroll
  for (int s = 256; s > 0; s >>= 1) {
    if (t < s) rbuf[t] += rbuf[t + s];
    __syncthreads();
  }
  float r = rbuf[0]; __syncthreads();
  return r;
}

// ---------------------------------------------------------------------------
// Kernel 1: build the 5 views (bf16 A-matrix, rows = view*64 + b)
// ---------------------------------------------------------------------------
__global__ __launch_bounds__(512) void build_views(
    const float* __restrict__ x, const float* __restrict__ n1,
    const float* __restrict__ n2, const int* __restrict__ fsp,
    const int* __restrict__ tsp, unsigned short* __restrict__ A) {
  __shared__ float re[4096];
  __shared__ float im[4096];
  __shared__ float rbuf[512];

  const int t = threadIdx.x;
  const int bidx = blockIdx.x >> 6;
  const int cidx = blockIdx.x & 63;
  const size_t rowoff = (size_t)blockIdx.x * T_SZ;
  const float* xr  = x  + rowoff;
  const float* n1r = n1 + rowoff;
  const float* n2r = n2 + rowoff;
  const size_t abase = (size_t)bidx * K_TOT + (size_t)cidx * T_SZ;
  const size_t VSTRIDE = (size_t)B_SZ * K_TOT;   // rows per view block

  float xo[8], sum = 0.f, ssq = 0.f;
  #pragma unroll
  for (int j = 0; j < 8; ++j) {
    const int tt = t + 512 * j;
    const float v = xr[tt];
    xo[j] = v; sum += v; ssq += v * v;
    re[tt] = v; im[tt] = 0.f;
  }
  const float tsum = blk_reduce_512(sum, rbuf);
  const float tssq = blk_reduce_512(ssq, rbuf);
  const float stdx =
      sqrtf(fmaxf((tssq - tsum * tsum * (1.f / 4096.f)) * (1.f / 4095.f), 0.f));

  // view 0 (original) and view 3 (noisy): no FFT needed
  #pragma unroll
  for (int j = 0; j < 8; ++j) {
    const int tt = t + 512 * j;
    A[abase + tt]               = f2bf(xo[j]);
    A[3 * VSTRIDE + abase + tt] = f2bf(xo[j] + n1r[tt] * (0.02f * stdx));
  }

  fft4096(re, im);                     // forward FFT of x
  float Xr[8], Xi[8];
  #pragma unroll
  for (int j = 0; j < 8; ++j) { Xr[j] = re[t + 512 * j]; Xi[j] = im[t + 512 * j]; }
  __syncthreads();

  const int fs = fsp[0], ts = tsp[0];
  const int fw = 409;   // int(0.1*4096)
  const int tw = 204;   // int(0.05*4096)
  const float invN = 1.f / 4096.f;

  // ---- view 1: compressed (keep k < 2048); ifft via conj trick ----
  #pragma unroll
  for (int j = 0; j < 8; ++j) {
    const int k = t + 512 * j;
    const float m = (k < 2048) ? 1.f : 0.f;
    re[k] = m * Xr[j]; im[k] = -m * Xi[j];
  }
  fft4096(re, im);
  #pragma unroll
  for (int j = 0; j < 8; ++j) {
    const int tt = t + 512 * j;
    A[VSTRIDE + abase + tt] = f2bf(re[tt] * invN);
  }
  __syncthreads();

  // ---- view 2: distorted (freq mask 0.1, then time mask 0.1) ----
  #pragma unroll
  for (int j = 0; j < 8; ++j) {
    const int k = t + 512 * j;
    const float m = (k >= fs && k < fs + fw) ? 0.1f : 1.f;
    re[k] = m * Xr[j]; im[k] = -m * Xi[j];
  }
  fft4096(re, im);
  #pragma unroll
  for (int j = 0; j < 8; ++j) {
    const int tt = t + 512 * j;
    const float tm = (tt >= ts && tt < ts + tw) ? 0.1f : 1.f;
    A[2 * VSTRIDE + abase + tt] = f2bf(re[tt] * invN * tm);
  }
  __syncthreads();

  // ---- view 4: combined = distort(compress(x,0.25)) + noise2 ----
  #pragma unroll
  for (int j = 0; j < 8; ++j) {
    const int k = t + 512 * j;
    float m = (k < 3072) ? 1.f : 0.f;                 // cutoff int(4096*0.75)
    m *= (k >= fs && k < fs + fw) ? 0.1f : 1.f;
    re[k] = m * Xr[j]; im[k] = -m * Xi[j];
  }
  fft4096(re, im);
  float y[8]; sum = 0.f; ssq = 0.f;
  #pragma unroll
  for (int j = 0; j < 8; ++j) {
    const int tt = t + 512 * j;
    const float tm = (tt >= ts && tt < ts + tw) ? 0.1f : 1.f;
    const float v = re[tt] * invN * tm;
    y[j] = v; sum += v; ssq += v * v;
  }
  const float ys  = blk_reduce_512(sum, rbuf);
  const float yss = blk_reduce_512(ssq, rbuf);
  const float stdy =
      sqrtf(fmaxf((yss - ys * ys * (1.f / 4096.f)) * (1.f / 4095.f), 0.f));
  #pragma unroll
  for (int j = 0; j < 8; ++j) {
    const int tt = t + 512 * j;
    A[4 * VSTRIDE + abase + tt] = f2bf(y[j] + n2r[tt] * (0.02f * stdy));
  }
}

// ---------------------------------------------------------------------------
// Kernel 2: split-K bf16 WMMA GEMM, A tile staged via Tensor Data Mover.
// Block = 320 thr (10 waves). Wave w owns tiles (mt = 2w + r, nt),
// r in {0,1}, nt in {0..3}: all fragment indices compile-time.
// ---------------------------------------------------------------------------
__global__ __launch_bounds__(320) void gemm_views(
    const unsigned short* __restrict__ A, const float* __restrict__ W,
    float* __restrict__ partials) {
  __shared__ __align__(16) unsigned short la[M_TOT * LA_STRIDE];   // ~25.6KB
  __shared__ __align__(16) unsigned short lb[N_TILE * LB_STRIDE];  // ~5KB [n][k]

  const int tt   = threadIdx.x;
  const int nb   = blockIdx.x;          // 0..7
  const int kc   = blockIdx.y;          // 0..63
  const int n0   = nb * N_TILE;
  const int k0   = kc * K_CHUNK;
  const int lane = tt & 31;
  const int wave = tt >> 5;             // 0..9
  const int m    = lane & 15;
  const int h    = lane >> 4;
  const int mt0  = wave * 2;            // wave's first M-tile

  // --- TDM descriptor group 1 (constant over the loop) ---
  // 2D tile: tile_dim0 = 32 elems (2B), tile_dim1 = 320 rows,
  // tensor_dim0(_stride) = K_TOT. LDS pad: every 16 dwords insert 4 dwords
  // (pad_interval=3, pad_amount=3) => 40-short padded row stride.
  i32x8 g1;
  g1[0] = (1 << 16) | (1 << 20) | (3 << 22) | (3 << 25); // data_size=2B, pad
  g1[1] = (int)((K_TOT & 0xFFFF) << 16);                  // tensor_dim0 lo16
  g1[2] = (int)((K_TOT >> 16) | (M_TOT << 16));           // dim0 hi / dim1 lo
  g1[3] = (int)(K_STEP << 16);                            // dim1 hi / tile_dim0
  g1[4] = M_TOT;                                          // tile_dim1, tile_dim2=0
  g1[5] = K_TOT;                                          // tensor_dim0_stride lo32
  g1[6] = 0;                                              // stride hi / dim1_stride lo
  g1[7] = 0;
  const i32x4 z4 = {0, 0, 0, 0};
  const unsigned int lds_base = (unsigned int)(unsigned long long)(void*)la;

  const v8f vzero = {0.f, 0.f, 0.f, 0.f, 0.f, 0.f, 0.f, 0.f};
  v8f acc[2][4];
  #pragma unroll
  for (int r = 0; r < 2; ++r)
    #pragma unroll
    for (int n = 0; n < 4; ++n) acc[r][n] = vzero;

  for (int ks = 0; ks < K_CHUNK; ks += K_STEP) {
    const int kb = k0 + ks;

    // --- A tile [320][32] bf16 -> LDS via Tensor Data Mover (wave 0) ---
    if (wave == 0) {
      const unsigned long long ga =
          (unsigned long long)A + (unsigned long long)kb * 2ull;
      u32x4 g0;
      g0[0] = 1u;                                  // count=1 (valid D#)
      g0[1] = lds_base;                            // lds_addr
      g0[2] = (unsigned int)ga;                    // global_addr[31:0]
      g0[3] = (unsigned int)((ga >> 32) & 0x1FFFFFFull) | (2u << 30); // type=2
#if defined(__clang_major__) && (__clang_major__ >= 23)
      const i32x8 z8 = {0, 0, 0, 0, 0, 0, 0, 0};
      __builtin_amdgcn_tensor_load_to_lds(g0, g1, z4, z4, z8, 0);
#else
      __builtin_amdgcn_tensor_load_to_lds(g0, g1, z4, z4, 0);
#endif
      __builtin_amdgcn_s_wait_tensorcnt(0);
    }

    // --- stage W tile 32x64 fp32 -> bf16, transposed into lb[n][k] ---
    #pragma unroll
    for (int i = 0; i < 2; ++i) {
      const int li = i * 320 + tt;      // 0..511 valid
      if (li < 512) {
        const int r  = li >> 4;         // k within step
        const int cq = li & 15;         // n/4
        const float* wp = W + (size_t)(kb + r) * FEAT + n0 + cq * 4;
        const float4 w = *(const float4*)wp;
        if (i == 0) __builtin_prefetch(wp + K_STEP * FEAT, 0, 1);
        lb[(cq * 4 + 0) * LB_STRIDE + r] = f2bf(w.x);
        lb[(cq * 4 + 1) * LB_STRIDE + r] = f2bf(w.y);
        lb[(cq * 4 + 2) * LB_STRIDE + r] = f2bf(w.z);
        lb[(cq * 4 + 3) * LB_STRIDE + r] = f2bf(w.w);
      }
    }
    __syncthreads();

    v16bf af[2], bf[4];
    #pragma unroll
    for (int r = 0; r < 2; ++r)
      af[r] = load_frag(la + ((mt0 + r) * 16 + m) * LA_STRIDE, h);
    #pragma unroll
    for (int n = 0; n < 4; ++n)
      bf[n] = load_frag(lb + (n * 16 + m) * LB_STRIDE, h);

    #pragma unroll
    for (int r = 0; r < 2; ++r)
      #pragma unroll
      for (int n = 0; n < 4; ++n)
        acc[r][n] = __builtin_amdgcn_wmma_f32_16x16x32_bf16(
            false, af[r], false, bf[n], (short)0, acc[r][n], false, false);
    __syncthreads();
  }

  // write partials[kc][mrow][ncol]
  #pragma unroll
  for (int r = 0; r < 2; ++r) {
    #pragma unroll
    for (int n = 0; n < 4; ++n) {
      const int ncol = n0 + n * 16 + m;
      #pragma unroll
      for (int e = 0; e < 8; ++e) {
        const int mrow = (mt0 + r) * 16 + e + 8 * h;
        partials[((size_t)kc * M_TOT + mrow) * FEAT + ncol] = acc[r][n][e];
      }
    }
  }
}

// ---------------------------------------------------------------------------
// Kernel 3: reduce split-K partials + bias -> allf[320][512]
// ---------------------------------------------------------------------------
__global__ __launch_bounds__(256) void reduce_partials(
    const float* __restrict__ partials, const float* __restrict__ bias,
    float* __restrict__ allf) {
  const int idx = blockIdx.x * 256 + threadIdx.x;  // 0..163839
  float s = 0.f;
  for (int kc = 0; kc < K_CHUNKS; ++kc)
    s += partials[(size_t)kc * (M_TOT * FEAT) + idx];
  allf[idx] = s + bias[idx & (FEAT - 1)];
}

// ---------------------------------------------------------------------------
// Kernel 4: consistency = mean_v mean_{b,d} (f0 - fv)^2
// ---------------------------------------------------------------------------
__global__ __launch_bounds__(512) void consistency_k(
    const float* __restrict__ allf, float* __restrict__ outs) {
  __shared__ float rbuf[512];
  const int t = threadIdx.x;
  float s = 0.f;
  for (int e = t; e < 4 * B_SZ * FEAT; e += 512) {
    const int v  = e >> 15;
    const int bd = e & 32767;
    const float d = allf[(size_t)(v + 1) * 32768 + bd] - allf[bd];
    s += d * d;
  }
  rbuf[t] = s; __syncthreads();
  #pragma unroll
  for (int st = 256; st > 0; st >>= 1) {
    if (t < st) rbuf[t] += rbuf[t + st];
    __syncthreads();
  }
  if (t == 0) outs[0] = rbuf[0] * (1.0f / (4.0f * 32768.0f));
}

// ---------------------------------------------------------------------------
// Kernel 5: L2-normalize feature rows
// ---------------------------------------------------------------------------
__global__ __launch_bounds__(128) void rownorm_k(
    const float* __restrict__ allf, float* __restrict__ fn) {
  __shared__ float rbuf[128];
  const int i = blockIdx.x, t = threadIdx.x;
  float s = 0.f;
  for (int d = t; d < FEAT; d += 128) {
    const float v = allf[(size_t)i * FEAT + d];
    s += v * v;
  }
  rbuf[t] = s; __syncthreads();
  #pragma unroll
  for (int st = 64; st > 0; st >>= 1) {
    if (t < st) rbuf[t] += rbuf[t + st];
    __syncthreads();
  }
  const float inv = 1.0f / sqrtf(rbuf[0]);
  for (int d = t; d < FEAT; d += 128)
    fn[(size_t)i * FEAT + d] = allf[(size_t)i * FEAT + d] * inv;
}

// ---------------------------------------------------------------------------
// Kernel 6: per-row sim + logsumexp + adjacent-mask contribution
// ---------------------------------------------------------------------------
__global__ __launch_bounds__(128) void simrow_k(
    const float* __restrict__ fn, float* __restrict__ rowc) {
  __shared__ float fni[FEAT];
  __shared__ float simr[M_TOT];
  __shared__ float rbuf[128];
  const int i = blockIdx.x, t = threadIdx.x;
  for (int d = t; d < FEAT; d += 128) fni[d] = fn[(size_t)i * FEAT + d];
  __syncthreads();
  for (int j = t; j < M_TOT; j += 128) {
    const float* fj = fn + (size_t)j * FEAT;
    float s = 0.f;
    for (int d = 0; d < FEAT; ++d) s += fni[d] * fj[d];
    simr[j] = s * 10.0f;            // / TEMPERATURE
  }
  __syncthreads();
  float mx = -1e30f;
  for (int j = t; j < M_TOT; j += 128) mx = fmaxf(mx, simr[j]);
  rbuf[t] = mx; __syncthreads();
  #pragma unroll
  for (int st = 64; st > 0; st >>= 1) {
    if (t < st) rbuf[t] = fmaxf(rbuf[t], rbuf[t + st]);
    __syncthreads();
  }
  mx = rbuf[0]; __syncthreads();
  float se = 0.f;
  for (int j = t; j < M_TOT; j += 128) se += __expf(simr[j] - mx);
  rbuf[t] = se; __syncthreads();
  #pragma unroll
  for (int st = 64; st > 0; st >>= 1) {
    if (t < st) rbuf[t] += rbuf[t + st];
    __syncthreads();
  }
  if (t == 0) {
    const float lse = mx + __logf(rbuf[0]);
    float c = 0.f; int cnt = 0;
    if (i > 0)         { c += simr[i - 1]; ++cnt; }
    if (i < M_TOT - 1) { c += simr[i + 1]; ++cnt; }
    rowc[i] = c - lse * (float)cnt;
  }
}

// ---------------------------------------------------------------------------
// Kernel 7: total = consistency + 0.5 * (-sum(rowc)/638)
// ---------------------------------------------------------------------------
__global__ __launch_bounds__(512) void finalize_k(
    const float* __restrict__ rowc, const float* __restrict__ consis,
    float* __restrict__ out) {
  __shared__ float rbuf[512];
  const int t = threadIdx.x;
  rbuf[t] = (t < M_TOT) ? rowc[t] : 0.f;
  __syncthreads();
  #pragma unroll
  for (int st = 256; st > 0; st >>= 1) {
    if (t < st) rbuf[t] += rbuf[t + st];
    __syncthreads();
  }
  if (t == 0) {
    const float contrastive = -rbuf[0] / 638.0f;   // sum(mask)=2*319
    out[0] = consis[0] + 0.5f * contrastive;
  }
}

// ---------------------------------------------------------------------------
// Workspace layout (bytes), ~211 MB total
// ---------------------------------------------------------------------------
static const size_t OFF_A    = 0;                                  // 320*262144*2
static const size_t OFF_PART = 167772160;                          // 64*320*512*4
static const size_t OFF_ALLF = OFF_PART + 41943040;                // 320*512*4
static const size_t OFF_FN   = OFF_ALLF + 655360;                  // 320*512*4
static const size_t OFF_RC   = OFF_FN + 655360;                    // 320*4
static const size_t OFF_CONS = OFF_RC + 2048;                      // 4

extern "C" void kernel_launch(void* const* d_in, const int* in_sizes, int n_in,
                              void* d_out, int out_size, void* d_ws,
                              size_t ws_size, hipStream_t stream) {
  const float* x  = (const float*)d_in[0];
  const float* W  = (const float*)d_in[1];
  const float* bv = (const float*)d_in[2];
  const float* n1 = (const float*)d_in[3];
  const float* n2 = (const float*)d_in[4];
  const int* fs   = (const int*)d_in[5];
  const int* ts   = (const int*)d_in[6];

  char* ws = (char*)d_ws;
  unsigned short* A = (unsigned short*)(ws + OFF_A);
  float* partials   = (float*)(ws + OFF_PART);
  float* allf       = (float*)(ws + OFF_ALLF);
  float* fn         = (float*)(ws + OFF_FN);
  float* rowc       = (float*)(ws + OFF_RC);
  float* consis     = (float*)(ws + OFF_CONS);
  float* out        = (float*)d_out;

  build_views<<<B_SZ * C_SZ, 512, 0, stream>>>(x, n1, n2, fs, ts, A);
  gemm_views<<<dim3(N_BLKS, K_CHUNKS), 320, 0, stream>>>(A, W, partials);
  reduce_partials<<<(M_TOT * FEAT) / 256, 256, 0, stream>>>(partials, bv, allf);
  consistency_k<<<1, 512, 0, stream>>>(allf, consis);
  rownorm_k<<<M_TOT, 128, 0, stream>>>(allf, fn);
  simrow_k<<<M_TOT, 128, 0, stream>>>(fn, rowc);
  finalize_k<<<1, 512, 0, stream>>>(rowc, consis, out);
}
